// NewMamba_5798205849759
// MI455X (gfx1250) — matrix operand order repaired and checked by reading, not compiled, not run on hardware
//
#include <hip/hip_runtime.h>
#include <hip/hip_bf16.h>

// Sizes from the reference
#define Hd    256
#define Id    512
#define Sdim  16
#define Rdim  16
#define KC    4
#define NL    8
#define Bn    16
#define Ln    2048
#define Mrows (Bn*Ln)          // 32768
#define EPSI  1e-5f

typedef __attribute__((ext_vector_type(16))) __bf16 v16bf;
typedef __attribute__((ext_vector_type(8)))  __bf16 v8bf;
typedef __attribute__((ext_vector_type(8)))  float  v8f;

__device__ __forceinline__ float sigmoidf_(float x) { return 1.f / (1.f + __expf(-x)); }

// ---------------- weight f32 -> bf16 conversion (once per launch) ----------------
__global__ void cvt_kernel(const float* __restrict__ src, __bf16* __restrict__ dst, int n) {
    int i = blockIdx.x * 256 + threadIdx.x;
    if (i < n) dst[i] = (__bf16)src[i];
}

// ---------------- RMSNorm -> bf16 activations ----------------
__global__ void rmsnorm_kernel(const float* __restrict__ x, const float* __restrict__ w,
                               __bf16* __restrict__ out) {
    __shared__ float red[256];
    int row = blockIdx.x;
    int t   = threadIdx.x;                 // H == 256 == blockDim.x
    float v = x[(size_t)row * Hd + t];
    red[t] = v * v;
    __syncthreads();
    for (int s = 128; s > 0; s >>= 1) {
        if (t < s) red[t] += red[t + s];
        __syncthreads();
    }
    float rs = rsqrtf(red[0] * (1.f / Hd) + EPSI);
    out[(size_t)row * Hd + t] = (__bf16)(v * rs * w[t]);
}

// ---------------- WMMA fragment helpers (v_wmma_f32_16x16x32_bf16) ----------------
// A fragment: row = lane&15; lanes 0-15 hold K [0..7]+[16..23], lanes 16-31 [8..15]+[24..31]
// B fragment: col = lane&15; lanes 0-15 hold K [0..7]+[16..23], lanes 16-31 [8..15]+[24..31]
__device__ __forceinline__ v16bf load_afrag(const __bf16* p) {
    v8bf a0 = *(const v8bf*)(p);
    v8bf a1 = *(const v8bf*)(p + 16);
    return __builtin_shufflevector(a0, a1, 0,1,2,3,4,5,6,7,8,9,10,11,12,13,14,15);
}
__device__ __forceinline__ v16bf load_bfrag(const __bf16* p) {
    v8bf b0 = *(const v8bf*)(p);
    v8bf b1 = *(const v8bf*)(p + 16);
    return __builtin_shufflevector(b0, b1, 0,1,2,3,4,5,6,7,8,9,10,11,12,13,14,15);
}
#define WMMA_BF16(acc, a, b) \
    __builtin_amdgcn_wmma_f32_16x16x32_bf16(false, (a), false, (b), (short)0, (acc), false, false)

// ---------------- 2x2 register-blocked bf16 GEMM ----------------
// Each wave computes a 32x32 output block (4 WMMA tiles) -> 2 b128 loads per WMMA.
// C[m,n] = sum_k A[m,k] * W[n,k]; optional fused residual (xsrc != nullptr).
__global__ void gemm_bf16_2x2_kernel(const __bf16* __restrict__ A, const __bf16* __restrict__ W,
                                     const float* __restrict__ xsrc, float* __restrict__ C,
                                     int Ntiles2, int Kd, int lda, int ldw, int ldc) {
    int lane = threadIdx.x & 31;
    int wave = (blockIdx.x * blockDim.x + threadIdx.x) >> 5;
    int tn = wave % Ntiles2;               // 32-wide N block index
    int tm = wave / Ntiles2;               // 32-tall M block index
    int ka = (lane & 16) ? 8 : 0;
    int kb = (lane & 16) ? 8 : 0;
    const __bf16* ap0 = A + (size_t)(tm * 32 + (lane & 15)) * lda + ka;
    const __bf16* ap1 = ap0 + (size_t)16 * lda;
    const __bf16* bp0 = W + (size_t)(tn * 32 + (lane & 15)) * ldw + kb;
    const __bf16* bp1 = bp0 + (size_t)16 * ldw;
    v8f acc00 = {0.f,0.f,0.f,0.f,0.f,0.f,0.f,0.f};
    v8f acc01 = acc00, acc10 = acc00, acc11 = acc00;
    for (int k0 = 0; k0 < Kd; k0 += 32) {
        v16bf a0 = load_afrag(ap0 + k0);
        v16bf a1 = load_afrag(ap1 + k0);
        v16bf b0 = load_bfrag(bp0 + k0);
        v16bf b1 = load_bfrag(bp1 + k0);
        acc00 = WMMA_BF16(acc00, a0, b0);
        acc01 = WMMA_BF16(acc01, a0, b1);
        acc10 = WMMA_BF16(acc10, a1, b0);
        acc11 = WMMA_BF16(acc11, a1, b1);
    }
    // D layout: VGPR r, lane l -> row r + (l>=16 ? 8 : 0), col l&15
    int crow = tm * 32 + ((lane >> 4) << 3);
    int ccol = tn * 32 + (lane & 15);
#pragma unroll
    for (int r = 0; r < 8; r++) {
        size_t i00 = (size_t)(crow + r) * ldc + ccol;
        size_t i10 = (size_t)(crow + 16 + r) * ldc + ccol;
        if (xsrc) {
            C[i00]      = xsrc[i00]      + acc00[r];
            C[i00 + 16] = xsrc[i00 + 16] + acc01[r];
            C[i10]      = xsrc[i10]      + acc10[r];
            C[i10 + 16] = xsrc[i10 + 16] + acc11[r];
        } else {
            C[i00]      = acc00[r];
            C[i00 + 16] = acc01[r];
            C[i10]      = acc10[r];
            C[i10 + 16] = acc11[r];
        }
    }
}

// ---------------- single-tile bf16 GEMM (for N = 48 x_proj) ----------------
__global__ void gemm_bf16_kernel(const __bf16* __restrict__ A, const __bf16* __restrict__ W,
                                 float* __restrict__ C, int Ntiles, int Kd,
                                 int lda, int ldw, int ldc) {
    int lane = threadIdx.x & 31;
    int wave = (blockIdx.x * blockDim.x + threadIdx.x) >> 5;
    int tn = wave % Ntiles;
    int tm = wave / Ntiles;
    int ka = (lane & 16) ? 8 : 0;
    int kb = (lane & 16) ? 8 : 0;
    const __bf16* ap = A + (size_t)(tm * 16 + (lane & 15)) * lda + ka;
    const __bf16* bp = W + (size_t)(tn * 16 + (lane & 15)) * ldw + kb;
    v8f acc = {0.f,0.f,0.f,0.f,0.f,0.f,0.f,0.f};
    for (int k0 = 0; k0 < Kd; k0 += 32) {
        v16bf af = load_afrag(ap + k0);
        v16bf bw = load_bfrag(bp + k0);
        acc = WMMA_BF16(acc, af, bw);
    }
    int crow = tm * 16 + ((lane >> 4) << 3);
    int ccol = tn * 16 + (lane & 15);
#pragma unroll
    for (int r = 0; r < 8; r++)
        C[(size_t)(crow + r) * ldc + ccol] = acc[r];
}

// ---------------- depthwise causal conv (K=4) + SiLU, dual f32/bf16 output ----------------
__global__ void conv_silu_kernel(const float* __restrict__ proj, const float* __restrict__ cw,
                                 const float* __restrict__ cb, float* __restrict__ uf,
                                 __bf16* __restrict__ ubf) {
    int idx = blockIdx.x * 256 + threadIdx.x;     // over Mrows * Id
    int i = idx & (Id - 1);
    int m = idx >> 9;                             // Id == 512
    int l = m & (Ln - 1);
    float s = cb[i];
    const float* wv = cw + i * KC;
#pragma unroll
    for (int j = 0; j < KC; j++) {
        int lj = l - (KC - 1) + j;
        if (lj >= 0)
            s = fmaf(proj[(size_t)(m - (KC - 1) + j) * (2 * Id) + i], wv[j], s);
    }
    float u = s * sigmoidf_(s);
    uf[idx]  = u;
    ubf[idx] = (__bf16)u;
}

// ---------------- fused selective scan ----------------
// one thread per (b, i) channel: dt_proj dot + softplus + recurrence + D skip + silu gate
__global__ void scan_kernel(const float* __restrict__ ssm, const float* __restrict__ uf,
                            const float* __restrict__ proj,   // gate at column Id + i
                            const float* __restrict__ dtw, const float* __restrict__ dtb,
                            const float* __restrict__ Alog, const float* __restrict__ Dv,
                            __bf16* __restrict__ ybf) {
    int b = blockIdx.x >> 1;
    int i = ((blockIdx.x & 1) << 8) + threadIdx.x;  // 0..511
    float w[Rdim], Ar[Sdim], st[Sdim];
#pragma unroll
    for (int r = 0; r < Rdim; r++) w[r] = dtw[(size_t)i * Rdim + r];
#pragma unroll
    for (int s = 0; s < Sdim; s++) { Ar[s] = -__expf(Alog[(size_t)i * Sdim + s]); st[s] = 0.f; }
    float bias  = dtb[i];
    float Dskip = Dv[i];
    for (int l = 0; l < Ln; l++) {
        size_t m = (size_t)b * Ln + l;
        const float* sr = ssm + m * 48;             // [0..15]=dt_in, [16..31]=B, [32..47]=C
        float dt = bias;
#pragma unroll
        for (int r = 0; r < Rdim; r++) dt = fmaf(sr[r], w[r], dt);
        dt = (dt > 20.f) ? dt : __logf(1.f + __expf(dt));   // softplus
        float ut  = uf[m * Id + i];
        float dtu = dt * ut;
        float y = 0.f;
#pragma unroll
        for (int s = 0; s < Sdim; s++) {
            float dA = __expf(dt * Ar[s]);
            st[s] = fmaf(dA, st[s], dtu * sr[16 + s]);
            y = fmaf(st[s], sr[32 + s], y);
        }
        float g = proj[m * (2 * Id) + Id + i];
        float o = (y + ut * Dskip) * (g * sigmoidf_(g));
        ybf[m * Id + i] = (__bf16)o;
    }
}

extern "C" void kernel_launch(void* const* d_in, const int* in_sizes, int n_in,
                              void* d_out, int out_size, void* d_ws, size_t ws_size,
                              hipStream_t stream) {
    const float* x     = (const float*)d_in[0];
    const float* normw = (const float*)d_in[1];
    const float* inw   = (const float*)d_in[2];
    const float* convw = (const float*)d_in[3];
    const float* convb = (const float*)d_in[4];
    const float* xpw   = (const float*)d_in[5];
    const float* dtw   = (const float*)d_in[6];
    const float* dtb   = (const float*)d_in[7];
    const float* alog  = (const float*)d_in[8];
    const float* Dv    = (const float*)d_in[9];
    const float* outw  = (const float*)d_in[10];
    float* out = (float*)d_out;

    // workspace carve-out (~332 MB total)
    char* base = (char*)d_ws;
    size_t off = 0;
    auto alloc = [&](size_t bytes) -> char* {
        char* r = base + off;
        off = (off + bytes + 255) & ~(size_t)255;
        return r;
    };
    float*  x_buf   = (float*) alloc((size_t)Mrows * Hd * 4);
    __bf16* h_bf    = (__bf16*)alloc((size_t)Mrows * Hd * 2);
    float*  proj    = (float*) alloc((size_t)Mrows * 2 * Id * 4);
    float*  u_f     = (float*) alloc((size_t)Mrows * Id * 4);
    __bf16* u_bf    = (__bf16*)alloc((size_t)Mrows * Id * 2);
    float*  ssm     = (float*) alloc((size_t)Mrows * 48 * 4);
    __bf16* y_bf    = (__bf16*)alloc((size_t)Mrows * Id * 2);
    __bf16* inw_bf  = (__bf16*)alloc((size_t)NL * 2 * Id * Hd * 2);
    __bf16* xpw_bf  = (__bf16*)alloc((size_t)NL * 48 * Id * 2);
    __bf16* outw_bf = (__bf16*)alloc((size_t)NL * Hd * Id * 2);
    (void)ws_size;

    // convert all layer weights to bf16 once
    int n1 = NL * 2 * Id * Hd;
    cvt_kernel<<<(n1 + 255) / 256, 256, 0, stream>>>(inw, inw_bf, n1);
    int n2 = NL * 48 * Id;
    cvt_kernel<<<(n2 + 255) / 256, 256, 0, stream>>>(xpw, xpw_bf, n2);
    int n3 = NL * Hd * Id;
    cvt_kernel<<<(n3 + 255) / 256, 256, 0, stream>>>(outw, outw_bf, n3);

    for (int li = 0; li < NL; li++) {
        const float* xsrc = (li == 0) ? x : x_buf;
        float* xdst = (li == NL - 1) ? out : x_buf;

        rmsnorm_kernel<<<Mrows, 256, 0, stream>>>(xsrc, normw + li * Hd, h_bf);

        {   // in_proj: M x (2I) = h_bf(M x H) @ in_w^T   (2x2 blocked)
            int Ntiles2 = (2 * Id) / 32;                 // 32
            int waves   = (Mrows / 32) * Ntiles2;        // 32768
            gemm_bf16_2x2_kernel<<<waves * 32 / 256, 256, 0, stream>>>(
                h_bf, inw_bf + (size_t)li * 2 * Id * Hd, nullptr, proj,
                Ntiles2, Hd, Hd, Hd, 2 * Id);
        }

        conv_silu_kernel<<<(size_t)Mrows * Id / 256, 256, 0, stream>>>(
            proj, convw + li * Id * KC, convb + li * Id, u_f, u_bf);

        {   // x_proj: M x 48 = u(M x I) @ xp_w^T   (single-tile, N=48)
            int Ntiles = 48 / 16;                        // 3
            int waves  = (Mrows / 16) * Ntiles;          // 6144
            gemm_bf16_kernel<<<waves * 32 / 256, 256, 0, stream>>>(
                u_bf, xpw_bf + (size_t)li * 48 * Id, ssm,
                Ntiles, Id, Id, Id, 48);
        }

        scan_kernel<<<Bn * 2, 256, 0, stream>>>(
            ssm, u_f, proj, dtw + (size_t)li * Id * Rdim, dtb + li * Id,
            alog + (size_t)li * Id * Sdim, Dv + li * Id, y_bf);

        {   // out_proj + residual: xdst = xsrc + y(M x I) @ out_w^T   (2x2 blocked)
            int Ntiles2 = Hd / 32;                       // 8
            int waves   = (Mrows / 32) * Ntiles2;        // 8192
            gemm_bf16_2x2_kernel<<<waves * 32 / 256, 256, 0, stream>>>(
                y_bf, outw_bf + (size_t)li * Hd * Id, xsrc, xdst,
                Ntiles2, Id, Id, Id, Hd);
        }
    }
    (void)in_sizes; (void)n_in; (void)out_size;
}
